// GATModel_20298015441203
// MI455X (gfx1250) — matrix operand order repaired and checked by reading, not compiled, not run on hardware
//
#include <hip/hip_runtime.h>
#include <hip/hip_bf16.h>
#include <float.h>

#define D_FEAT 128
#define LRELU_SLOPE 0.2f
#define SM_EPS 1e-16f

typedef __attribute__((ext_vector_type(16))) _Float16 v16h;
typedef __attribute__((ext_vector_type(8)))  _Float16 v8h;
typedef __attribute__((ext_vector_type(8)))  float    v8f;

// ---------- helpers ----------
__device__ __forceinline__ int f2key(float f) {
    int i = __float_as_int(f);
    return (i >= 0) ? i : (i ^ 0x7FFFFFFF);   // order-preserving int key
}
__device__ __forceinline__ float key2f(int k) {
    return __int_as_float((k >= 0) ? k : (k ^ 0x7FFFFFFF));
}
__device__ __forceinline__ float lrelu(float e) {
    return e > 0.f ? e : LRELU_SLOPE * e;
}
__device__ __forceinline__ void edge_sd(const int* __restrict__ ei, int e, int E,
                                        int& s, int& d) {
    if (e < E) { s = ei[2 * e]; d = ei[2 * e + 1]; }
    else       { s = e - E; d = e - E; }        // self loops appended
}

// ---------- embedding: h = x * W_emb + b_emb  (store f16 for WMMA) ----------
__global__ void embed_kernel(const float* __restrict__ x,
                             const float* __restrict__ wemb,
                             const float* __restrict__ bemb,
                             _Float16* __restrict__ h16, int total) {
    int idx = blockIdx.x * blockDim.x + threadIdx.x;
    if (idx >= total) return;
    int n = idx >> 7, c = idx & 127;
    h16[idx] = (_Float16)(x[n] * wemb[c] + bemb[c]);
}

// ---------- W[k][j] (row-major f32) -> Wt[j][k] f16 ----------
__global__ void convw_kernel(const float* __restrict__ W,
                             _Float16* __restrict__ Wt) {
    int idx = blockIdx.x * blockDim.x + threadIdx.x;   // 128*128 threads
    if (idx >= D_FEAT * D_FEAT) return;
    int k = idx >> 7, j = idx & 127;
    Wt[j * D_FEAT + k] = (_Float16)W[k * D_FEAT + j];
}

// ---------- WMMA GEMM: C[n,j] = sum_k A16[n,k] * W[k,j],  Bt = W^T f16 -------
__global__ void __launch_bounds__(256)
gat_gemm_wmma(const _Float16* __restrict__ A16,   // [nrows,128] row-major
              const _Float16* __restrict__ Bt16,  // [128,128] = W transposed
              float* __restrict__ Cout) {          // [nrows,128]
    const int lane = threadIdx.x & 31;
    const int wave = threadIdx.x >> 5;            // 8 waves -> 8 col tiles
    const int row0 = blockIdx.x << 4;
    const int col0 = wave << 4;
    const int m  = lane & 15;                     // row in A tile / col in B tile
    const int kg = lane >> 4;                     // K half-group select

    const _Float16* ap = A16 + (size_t)(row0 + m) * D_FEAT + kg * 8;
    const _Float16* bp = Bt16 + (size_t)(col0 + m) * D_FEAT + kg * 8;

    v8f acc = {};
#pragma unroll
    for (int kb = 0; kb < 4; ++kb) {
        union { v16h v; v8h h[2]; } a, b;
        // A 16x32 f16 layout: lanes<16 K={0..7,16..23}, lanes>=16 K={8..15,24..31}
        a.h[0] = *(const v8h*)(ap + kb * 32);
        a.h[1] = *(const v8h*)(ap + kb * 32 + 16);
        // B 32x16 f16 layout: lanes<16 K={0..7,16..23}, lanes>=16 K={8..15,24..31}, col = m
        b.h[0] = *(const v8h*)(bp + kb * 32);
        b.h[1] = *(const v8h*)(bp + kb * 32 + 16);
        acc = __builtin_amdgcn_wmma_f32_16x16x32_f16(
            /*neg_a=*/false, a.v, /*neg_b=*/false, b.v,
            /*c_mod=*/(short)0, acc, /*reuse_a=*/false, /*reuse_b=*/false);
    }
    // C f32 16x16: lane (m, kg), VGPR r -> element [M = r + kg*8, N = m]
    float* cp = Cout + (size_t)row0 * D_FEAT + col0 + m;
#pragma unroll
    for (int r = 0; r < 8; ++r)
        cp[(size_t)(r + kg * 8) * D_FEAT] = acc[r];
}

// ---------- per-node attention dots ----------
__global__ void alpha_kernel(const float* __restrict__ xp,
                             const float* __restrict__ a_src,
                             const float* __restrict__ a_dst,
                             float* __restrict__ asb, float* __restrict__ adb,
                             int n, int heads, int fh) {
    int idx = blockIdx.x * blockDim.x + threadIdx.x;
    if (idx >= n * heads) return;
    int node = idx / heads, h = idx - node * heads;
    const float* xr = xp + (size_t)node * D_FEAT + h * fh;
    const float* as = a_src + h * fh;
    const float* ad = a_dst + h * fh;
    float ss = 0.f, sd = 0.f;
    for (int f = 0; f < fh; ++f) { float v = xr[f]; ss += v * as[f]; sd += v * ad[f]; }
    asb[idx] = ss; adb[idx] = sd;
}

// ---------- per-layer state init ----------
__global__ void init_kernel(float* __restrict__ agg, int* __restrict__ mkey,
                            float* __restrict__ denom, int nd, int nh) {
    int idx = blockIdx.x * blockDim.x + threadIdx.x;
    if (idx < nd) agg[idx] = 0.f;
    if (idx < nh) { mkey[idx] = f2key(-FLT_MAX); denom[idx] = 0.f; }
}

// ---------- segment max over destination ----------
__global__ void edge_max_kernel(const int* __restrict__ ei,
                                const float* __restrict__ asb,
                                const float* __restrict__ adb,
                                int* __restrict__ mkey,
                                int E, int EE, int heads) {
    int idx = blockIdx.x * blockDim.x + threadIdx.x;
    if (idx >= EE * heads) return;
    int e = idx / heads, h = idx - e * heads;
    int s, d; edge_sd(ei, e, E, s, d);
    float ee = lrelu(asb[s * heads + h] + adb[d * heads + h]);
    atomicMax(&mkey[d * heads + h], f2key(ee));
}

// ---------- segment sum of exp ----------
__global__ void edge_sum_kernel(const int* __restrict__ ei,
                                const float* __restrict__ asb,
                                const float* __restrict__ adb,
                                const int* __restrict__ mkey,
                                float* __restrict__ denom,
                                int E, int EE, int heads) {
    int idx = blockIdx.x * blockDim.x + threadIdx.x;
    if (idx >= EE * heads) return;
    int e = idx / heads, h = idx - e * heads;
    int s, d; edge_sd(ei, e, E, s, d);
    int dh = d * heads + h;
    float ee = lrelu(asb[s * heads + h] + adb[dh]);
    atomicAdd(&denom[dh], __expf(ee - key2f(mkey[dh])));
}

// ---------- message aggregation: one wave per edge, float4 per lane ----------
__global__ void __launch_bounds__(256)
edge_agg_kernel(const int* __restrict__ ei,
                const float* __restrict__ asb, const float* __restrict__ adb,
                const int* __restrict__ mkey, const float* __restrict__ denom,
                const float* __restrict__ xp, float* __restrict__ agg,
                int E, int EE, int heads, int fh) {
    int e = blockIdx.x * (blockDim.x >> 5) + (threadIdx.x >> 5);
    if (e >= EE) return;
    int lane = threadIdx.x & 31;
    int s, d; edge_sd(ei, e, E, s, d);
    int j = lane * 4;                   // 4 consecutive features per lane
    int h = j / fh;                     // fh >= 4, aligned -> single head per lane
    int dh = d * heads + h;
    float ee   = lrelu(asb[s * heads + h] + adb[dh]);
    float a    = __expf(ee - key2f(mkey[dh]));
    float coef = a / (denom[dh] + SM_EPS);
    const float4 xv = *(const float4*)(xp + (size_t)s * D_FEAT + j);
    float* op = agg + (size_t)d * D_FEAT + j;
    atomicAdd(op + 0, coef * xv.x);
    atomicAdd(op + 1, coef * xv.y);
    atomicAdd(op + 2, coef * xv.z);
    atomicAdd(op + 3, coef * xv.w);
}

// ---------- bias + activation, emit f16 for next layer or f32 output ----------
__global__ void finalize_kernel(const float* __restrict__ agg,
                                const float* __restrict__ bias,
                                _Float16* __restrict__ h16,
                                float* __restrict__ fout,
                                int total, int relu) {
    int idx = blockIdx.x * blockDim.x + threadIdx.x;
    if (idx >= total) return;
    float v = agg[idx] + bias[idx & 127];
    if (relu) v = fmaxf(v, 0.f);
    if (fout) fout[idx] = v;
    if (h16)  h16[idx] = (_Float16)v;
}

extern "C" void kernel_launch(void* const* d_in, const int* in_sizes, int n_in,
                              void* d_out, int out_size, void* d_ws, size_t ws_size,
                              hipStream_t stream) {
    (void)n_in; (void)out_size; (void)ws_size;
    const float* x     = (const float*)d_in[0];
    const int*   eidx  = (const int*)d_in[1];
    const float* W_emb = (const float*)d_in[2];
    const float* b_emb = (const float*)d_in[3];
    const int N  = in_sizes[0];
    const int E  = in_sizes[1] / 2;
    const int EE = E + N;
    const size_t ND = (size_t)N * D_FEAT;

    struct Layer { const float *W, *as, *ad, *b; int heads, relu; };
    Layer L[4] = {
        {(const float*)d_in[4],  (const float*)d_in[5],  (const float*)d_in[6],  (const float*)d_in[7],  4, 1},
        {(const float*)d_in[8],  (const float*)d_in[9],  (const float*)d_in[10], (const float*)d_in[11], 1, 1},
        {(const float*)d_in[12], (const float*)d_in[13], (const float*)d_in[14], (const float*)d_in[15], 1, 1},
        {(const float*)d_in[16], (const float*)d_in[17], (const float*)d_in[18], (const float*)d_in[19], 1, 0},
    };

    // carve workspace (256B aligned chunks)
    char* p = (char*)d_ws;
    auto take = [&](size_t bytes) -> void* {
        void* q = (void*)p; p += (bytes + 255) & ~(size_t)255; return q;
    };
    _Float16* h16   = (_Float16*)take(ND * 2);                  // layer input, f16
    float*    xp    = (float*)take(ND * 4);                     // h @ W
    float*    agg   = (float*)take(ND * 4);                     // scatter accum
    float*    asb   = (float*)take((size_t)N * 4 * 4);          // alpha_src [N,H<=4]
    float*    adb   = (float*)take((size_t)N * 4 * 4);          // alpha_dst
    int*      mkey  = (int*)take((size_t)N * 4 * 4);            // seg-max keys
    float*    denom = (float*)take((size_t)N * 4 * 4);          // seg-sum
    _Float16* wt    = (_Float16*)take((size_t)D_FEAT * D_FEAT * 2); // W^T f16

    const int TB = 256;
    embed_kernel<<<(unsigned)((ND + TB - 1) / TB), TB, 0, stream>>>(
        x, W_emb, b_emb, h16, (int)ND);

    for (int li = 0; li < 4; ++li) {
        const int heads = L[li].heads, fh = D_FEAT / heads;
        const bool last = (li == 3);
        convw_kernel<<<(D_FEAT * D_FEAT + TB - 1) / TB, TB, 0, stream>>>(L[li].W, wt);
        gat_gemm_wmma<<<N / 16, TB, 0, stream>>>(h16, wt, xp);
        alpha_kernel<<<(N * heads + TB - 1) / TB, TB, 0, stream>>>(
            xp, L[li].as, L[li].ad, asb, adb, N, heads, fh);
        init_kernel<<<(unsigned)((ND + TB - 1) / TB), TB, 0, stream>>>(
            agg, mkey, denom, (int)ND, N * heads);
        edge_max_kernel<<<(EE * heads + TB - 1) / TB, TB, 0, stream>>>(
            eidx, asb, adb, mkey, E, EE, heads);
        edge_sum_kernel<<<(EE * heads + TB - 1) / TB, TB, 0, stream>>>(
            eidx, asb, adb, mkey, denom, E, EE, heads);
        edge_agg_kernel<<<(EE + 7) / 8, TB, 0, stream>>>(
            eidx, asb, adb, mkey, denom, xp, agg, E, EE, heads, fh);
        finalize_kernel<<<(unsigned)((ND + TB - 1) / TB), TB, 0, stream>>>(
            agg, L[li].b, last ? nullptr : h16,
            last ? (float*)d_out : nullptr, (int)ND, L[li].relu);
    }
}
